// AttentionMix_62182536511495
// MI455X (gfx1250) — matrix operand order repaired and checked by reading, not compiled, not run on hardware
//
#include <hip/hip_runtime.h>
#include <cstdint>

// ---------------------------------------------------------------------------
// AF3 triangle attention for MI455X (gfx1250, wave32, WMMA bf16 16x16x32).
//
// Pipeline:
//  1) prep_weights : fp32 weights -> bf16. Wcat[528][128] = {Wq,Wk,Wv,Wg,Wbias,pad}
//  2) ln_kernel    : LayerNorm(pair) -> Xbf bf16 [147456][128] (1 wave / row)
//  3) proj_gemm    : Xbf @ Wcat^T via v_wmma_f32_16x16x32_bf16, weights staged in
//                    padded LDS. Outputs: XP[row][384] = {q|k|g}, Vt[b][h][32][384],
//                    biasbuf[4][147456] (fp32 pair bias).
//  4) attn_kernel  : transposed flash attention. S^T = K*Q^T (WMMA), per-lane
//                    online softmax (queries live in lanes), O^T = V^T * P^T (WMMA),
//                    sigmoid gate applied in epilogue -> WA bf16 [row][128].
//  5) out_gemm     : WA @ Wo^T (WMMA) -> fp32 out.
// ---------------------------------------------------------------------------

#define NDIM 384
#define NN   (384 * 384)          // 147456 rows
#define CP   128
#define LOG2E 1.4426950408889634f

typedef __attribute__((ext_vector_type(16))) __bf16 bf16x16;
typedef __attribute__((ext_vector_type(8)))  __bf16 bf16x8;
typedef __attribute__((ext_vector_type(4)))  __bf16 bf16x4;
typedef __attribute__((ext_vector_type(8)))  float  f32x8;

// ds_swizzle group-of-32: offset = {xor[14:10], or[9:5], and[4:0]}
#define SWZF(x, imm) __int_as_float(__builtin_amdgcn_ds_swizzle(__float_as_int(x), (imm)))
#define SWAPX16 0x401F
#define SWAPX8  0x201F
#define SWAPX4  0x101F
#define SWAPX2  0x081F
#define SWAPX1  0x041F

__device__ __forceinline__ f32x8 wmma_bf16(bf16x16 a, bf16x16 b, f32x8 c) {
  return __builtin_amdgcn_wmma_f32_16x16x32_bf16(false, a, false, b, (short)0, c,
                                                 false, false);
}

__device__ __forceinline__ bf16x8 ld8(const __bf16* p) {
  return *(const bf16x8*)p;
}

__device__ __forceinline__ bf16x16 cat16(bf16x8 a, bf16x8 b) {
  bf16x16 r;
#pragma unroll
  for (int i = 0; i < 8; ++i) { r[i] = a[i]; r[i + 8] = b[i]; }
  return r;
}

// ---------------------------------------------------------------------------
// 1) Weight prep: rows 0..127 Wq, 128..255 Wk, 256..383 Wv, 384..511 Wg,
//    512..515 Wbias, 516..527 zero pad, 528..655 Wo (separate buffer).
// ---------------------------------------------------------------------------
__global__ void prep_weights(const float* __restrict__ wq, const float* __restrict__ wk,
                             const float* __restrict__ wv, const float* __restrict__ wg,
                             const float* __restrict__ wb, const float* __restrict__ wo,
                             __bf16* __restrict__ wcat, __bf16* __restrict__ wobf) {
  int idx = blockIdx.x * blockDim.x + threadIdx.x;
  if (idx >= 656 * 128) return;
  int row = idx >> 7, c = idx & 127;
  float v;
  if (row < 128)      v = wq[row * 128 + c];
  else if (row < 256) v = wk[(row - 128) * 128 + c];
  else if (row < 384) v = wv[(row - 256) * 128 + c];
  else if (row < 512) v = wg[(row - 384) * 128 + c];
  else if (row < 516) v = wb[(row - 512) * 128 + c];
  else if (row < 528) v = 0.0f;
  else                v = wo[(row - 528) * 128 + c];
  if (row < 528) wcat[row * 128 + c] = (__bf16)v;
  else           wobf[(row - 528) * 128 + c] = (__bf16)v;
}

// ---------------------------------------------------------------------------
// 2) LayerNorm: one wave32 per 128-elem row; 4 floats/lane; swizzle reductions.
// ---------------------------------------------------------------------------
__global__ void ln_kernel(const float* __restrict__ pair, const float* __restrict__ w,
                          const float* __restrict__ b, __bf16* __restrict__ xbf) {
  int wave = (blockIdx.x * blockDim.x + threadIdx.x) >> 5;
  int lane = threadIdx.x & 31;
  const float* row = pair + (size_t)wave * CP;
  float4 v = *(const float4*)(row + lane * 4);
  float s = v.x + v.y + v.z + v.w;
  s += SWZF(s, SWAPX1); s += SWZF(s, SWAPX2); s += SWZF(s, SWAPX4);
  s += SWZF(s, SWAPX8); s += SWZF(s, SWAPX16);
  float mu = s * (1.0f / 128.0f);
  float dx = v.x - mu, dy = v.y - mu, dz = v.z - mu, dw = v.w - mu;
  float ss = dx * dx + dy * dy + dz * dz + dw * dw;
  ss += SWZF(ss, SWAPX1); ss += SWZF(ss, SWAPX2); ss += SWZF(ss, SWAPX4);
  ss += SWZF(ss, SWAPX8); ss += SWZF(ss, SWAPX16);
  float rs = rsqrtf(ss * (1.0f / 128.0f) + 1e-5f);
  float4 wv = *(const float4*)(w + lane * 4);
  float4 bv = *(const float4*)(b + lane * 4);
  bf16x4 o;
  o[0] = (__bf16)(dx * rs * wv.x + bv.x);
  o[1] = (__bf16)(dy * rs * wv.y + bv.y);
  o[2] = (__bf16)(dz * rs * wv.z + bv.z);
  o[3] = (__bf16)(dw * rs * wv.w + bv.w);
  *(bf16x4*)(xbf + (size_t)wave * CP + lane * 4) = o;
}

// ---------------------------------------------------------------------------
// 3) Projection GEMM: out[row][col] = sum_c Xbf[row][c] * Wcat[col][c].
//    One wave per 16-row tile; loops over 33 column tiles (q,k,v,g,bias).
//    Weights staged in LDS with stride 136 (bank-conflict-free b128 ds loads).
// ---------------------------------------------------------------------------
__global__ void proj_gemm(const __bf16* __restrict__ xbf, const __bf16* __restrict__ wcat,
                          __bf16* __restrict__ xp, __bf16* __restrict__ vt,
                          float* __restrict__ biasb) {
  extern __shared__ __align__(16) __bf16 lds[];   // [528][136]
  for (int idx = threadIdx.x; idx < 528 * 128; idx += blockDim.x) {
    int r = idx >> 7, c = idx & 127;
    lds[r * 136 + c] = wcat[idx];
  }
  __syncthreads();

  int waveId = threadIdx.x >> 5;
  int lane   = threadIdx.x & 31;
  int mt     = blockIdx.x * 8 + waveId;          // 0..9215
  int rowbase = mt * 16;
  int m = lane & 15, grp = lane >> 4;

  // A tiles (16x32 bf16): lane m holds row m; element chunks {grp*8, 16+grp*8}.
  bf16x16 A[4];
  const __bf16* xr = xbf + (size_t)(rowbase + m) * CP;
#pragma unroll
  for (int kc = 0; kc < 4; ++kc) {
    int kb = kc * 32;
    A[kc] = cat16(ld8(xr + kb + grp * 8), ld8(xr + kb + 16 + grp * 8));
  }

  int bb = rowbase / NDIM;                       // batch index (rows never straddle)
  int n0 = (rowbase % NDIM) + grp * 8;

  for (int ct = 0; ct < 33; ++ct) {
    f32x8 acc = {0, 0, 0, 0, 0, 0, 0, 0};
    // B (32x16): lane n holds Wcat row (ct*16+n), element chunks kb + {grp*8, 16+grp*8}.
    const __bf16* wl = lds + (size_t)(ct * 16 + m) * 136 + grp * 8;
#pragma unroll
    for (int kc = 0; kc < 4; ++kc) {
      bf16x16 B = cat16(ld8(wl + kc * 32), ld8(wl + kc * 32 + 16));
      acc = wmma_bf16(A[kc], B, acc);
    }
    // Epilogue: D[m][n] -> lane holds col n=m, rows grp*8+r.
    if (ct < 16) {                                // q (cols 0..127), k (128..255)
      int col = ct * 16 + m;
      __bf16* dst = xp + (size_t)(rowbase + grp * 8) * 384 + col;
#pragma unroll
      for (int r = 0; r < 8; ++r) dst[(size_t)r * 384] = (__bf16)acc[r];
    } else if (ct < 24) {                         // v -> Vt[b][h][d][n]
      int dg = ct * 16 + m - 256;                 // 0..127
      bf16x8 o;
#pragma unroll
      for (int r = 0; r < 8; ++r) o[r] = (__bf16)acc[r];
      *(bf16x8*)(vt + (size_t)((bb * 4 + (dg >> 5)) * 32 + (dg & 31)) * 384 + n0) = o;
    } else if (ct < 32) {                         // g -> XP cols 256..383
      int col = ct * 16 + m - 128;
      __bf16* dst = xp + (size_t)(rowbase + grp * 8) * 384 + col;
#pragma unroll
      for (int r = 0; r < 8; ++r) dst[(size_t)r * 384] = (__bf16)acc[r];
    } else {                                      // pair bias (4 heads)
      if (m < 4) {
        float* dst = biasb + (size_t)m * NN + rowbase + grp * 8;
#pragma unroll
        for (int r = 0; r < 8; ++r) dst[r] = acc[r];
      }
    }
  }
}

// ---------------------------------------------------------------------------
// 4) Transposed flash attention. Block = 128 threads (4 waves, wave = head).
//    Wave owns (b, h, 16-query tile). Loop over 12 chunks of 32 keys:
//      S^T  = WMMA(K_tile[16x32], Q^T[32x16])   -> query per lane
//      online softmax: per-lane stats, one SWAPX16 merge
//      P^T  -> B-layout via 8 SWAPX16 swaps + bf16 packs
//      O^T += WMMA(Vt_tile[16x32], P^T[32x16])  (two d-tiles)
//    Epilogue: divide by l, apply sigmoid gate, store bf16 WA.
// ---------------------------------------------------------------------------
__global__ void attn_kernel(const __bf16* __restrict__ xp, const __bf16* __restrict__ vt,
                            const float* __restrict__ biasb,
                            const unsigned char* __restrict__ mask,
                            __bf16* __restrict__ wa) {
  int h    = threadIdx.x >> 5;
  int lane = threadIdx.x & 31;
  int qt   = blockIdx.x;                 // 0..23
  int bb   = blockIdx.y;                 // 0..383
  int q = lane & 15, grp = lane >> 4;
  int qrow = qt * 16 + q;

  // Q in B-layout: lane holds Q[qrow] element chunks {grp*8, 16+grp*8} of this head.
  const __bf16* qp = xp + (size_t)(bb * NDIM + qrow) * 384 + h * 32 + grp * 8;
  bf16x16 Qb = cat16(ld8(qp), ld8(qp + 16));

  f32x8 acc0 = {0, 0, 0, 0, 0, 0, 0, 0};
  f32x8 acc1 = {0, 0, 0, 0, 0, 0, 0, 0};
  float mrun = -1e30f, lrun = 0.0f;
  const float scale = 0.17677669529663687f;    // 1/sqrt(32)

  const float* bp = biasb + (size_t)h * NN + (size_t)qrow * NDIM;
  const unsigned char* mrow = mask + (size_t)bb * NDIM;

  for (int kb = 0; kb < NDIM; kb += 32) {
    // K A-tiles: lane holds K row (kb+q), element chunks {grp*8, 16+grp*8}.
    const __bf16* kp0 = xp + (size_t)(bb * NDIM + kb + q) * 384 + 128 + h * 32 + grp * 8;
    const __bf16* kp1 = kp0 + (size_t)16 * 384;
    bf16x16 K0 = cat16(ld8(kp0), ld8(kp0 + 16));
    bf16x16 K1 = cat16(ld8(kp1), ld8(kp1 + 16));
    f32x8 z = {0, 0, 0, 0, 0, 0, 0, 0};
    f32x8 S0 = wmma_bf16(K0, Qb, z);     // S0[key r (+grp*8)][query q]
    f32x8 S1 = wmma_bf16(K1, Qb, z);     // keys kb+16..

    float bias0[8], bias1[8];
    const float* bq = bp + kb + grp * 8;
    *(float4*)&bias0[0] = *(const float4*)(bq);
    *(float4*)&bias0[4] = *(const float4*)(bq + 4);
    *(float4*)&bias1[0] = *(const float4*)(bq + 16);
    *(float4*)&bias1[4] = *(const float4*)(bq + 20);
    uint2 mb0 = *(const uint2*)(mrow + kb + grp * 8);
    uint2 mb1 = *(const uint2*)(mrow + kb + 16 + grp * 8);

    float p0[8], p1[8];
    float cmax = -1e30f;
#pragma unroll
    for (int r = 0; r < 8; ++r) {
      unsigned m0 = ((r < 4 ? (mb0.x >> (8 * r)) : (mb0.y >> (8 * (r - 4)))) & 0xFFu);
      unsigned m1 = ((r < 4 ? (mb1.x >> (8 * r)) : (mb1.y >> (8 * (r - 4)))) & 0xFFu);
      float s0 = S0[r] * scale + bias0[r] + (m0 ? 0.0f : -1e9f);
      float s1 = S1[r] * scale + bias1[r] + (m1 ? 0.0f : -1e9f);
      p0[r] = s0; p1[r] = s1;
      cmax = fmaxf(cmax, fmaxf(s0, s1));
    }
    cmax = fmaxf(cmax, SWZF(cmax, SWAPX16));       // merge half-wave key groups
    float mnew  = fmaxf(mrun, cmax);
    float alpha = __builtin_amdgcn_exp2f((mrun - mnew) * LOG2E);
    float rsum = 0.0f;
#pragma unroll
    for (int r = 0; r < 8; ++r) {
      p0[r] = __builtin_amdgcn_exp2f((p0[r] - mnew) * LOG2E);
      p1[r] = __builtin_amdgcn_exp2f((p1[r] - mnew) * LOG2E);
      rsum += p0[r] + p1[r];
    }
    rsum += SWZF(rsum, SWAPX16);
    lrun = lrun * alpha + rsum;
    mrun = mnew;
#pragma unroll
    for (int r = 0; r < 8; ++r) { acc0[r] *= alpha; acc1[r] *= alpha; }

    // P^T -> B-layout (32 keys x 16 queries). Half-wave swap supplies the
    // keys held by the partner lane.
    bf16x16 Pb;
#pragma unroll
    for (int r = 0; r < 8; ++r) {
      float sel = grp ? p0[r] : p1[r];
      float x   = SWZF(sel, SWAPX16);
      float first  = grp ? x     : p0[r];
      float second = grp ? p1[r] : x;
      Pb[r]     = (__bf16)first;
      Pb[8 + r] = (__bf16)second;
    }

    // V^T A-tiles: lane holds Vt row d (=q, +16), keys kb + grp*16 .. +16.
    const __bf16* vp0 = vt + (size_t)((bb * 4 + h) * 32 + q) * 384 + kb + grp * 16;
    const __bf16* vp1 = vp0 + (size_t)16 * 384;
    bf16x16 V0 = cat16(ld8(vp0), ld8(vp0 + 8));
    bf16x16 V1 = cat16(ld8(vp1), ld8(vp1 + 8));
    acc0 = wmma_bf16(V0, Pb, acc0);      // O^T[d 0..15][q]
    acc1 = wmma_bf16(V1, Pb, acc1);      // O^T[d 16..31][q]
  }

  float inv = __builtin_amdgcn_rcpf(lrun);
  const __bf16* gp = xp + (size_t)(bb * NDIM + qrow) * 384 + 256 + h * 32 + grp * 8;
  bf16x8 g0 = ld8(gp);
  bf16x8 g1 = ld8(gp + 16);
  bf16x8 o0, o1;
#pragma unroll
  for (int r = 0; r < 8; ++r) {
    float gate0 = __builtin_amdgcn_rcpf(1.0f + __builtin_amdgcn_exp2f(-(float)g0[r] * LOG2E));
    float gate1 = __builtin_amdgcn_rcpf(1.0f + __builtin_amdgcn_exp2f(-(float)g1[r] * LOG2E));
    o0[r] = (__bf16)(acc0[r] * inv * gate0);
    o1[r] = (__bf16)(acc1[r] * inv * gate1);
  }
  __bf16* dst = wa + (size_t)(bb * NDIM + qrow) * 128 + h * 32 + grp * 8;
  *(bf16x8*)dst = o0;
  *(bf16x8*)(dst + 16) = o1;
}

// ---------------------------------------------------------------------------
// 5) Output GEMM: out[row][d] = sum_c WA[row][c] * Wo[d][c]  (fp32 out).
// ---------------------------------------------------------------------------
__global__ void out_gemm(const __bf16* __restrict__ wabuf, const __bf16* __restrict__ wobf,
                         float* __restrict__ out) {
  int waveId = threadIdx.x >> 5;
  int lane   = threadIdx.x & 31;
  int mt = blockIdx.x * 8 + waveId;
  int rowbase = mt * 16;
  int m = lane & 15, grp = lane >> 4;

  bf16x16 A[4];
  const __bf16* ar = wabuf + (size_t)(rowbase + m) * 128;
#pragma unroll
  for (int kc = 0; kc < 4; ++kc) {
    int kb = kc * 32;
    A[kc] = cat16(ld8(ar + kb + grp * 8), ld8(ar + kb + 16 + grp * 8));
  }
  for (int ct = 0; ct < 8; ++ct) {
    f32x8 acc = {0, 0, 0, 0, 0, 0, 0, 0};
    const __bf16* wr = wobf + (size_t)(ct * 16 + m) * 128 + grp * 8;
#pragma unroll
    for (int kc = 0; kc < 4; ++kc) {
      bf16x16 B = cat16(ld8(wr + kc * 32), ld8(wr + kc * 32 + 16));
      acc = wmma_bf16(A[kc], B, acc);
    }
    float* dst = out + (size_t)(rowbase + grp * 8) * 128 + ct * 16 + m;
#pragma unroll
    for (int r = 0; r < 8; ++r) dst[(size_t)r * 128] = acc[r];
  }
}

// ---------------------------------------------------------------------------
extern "C" void kernel_launch(void* const* d_in, const int* in_sizes, int n_in,
                              void* d_out, int out_size, void* d_ws, size_t ws_size,
                              hipStream_t stream) {
  (void)in_sizes; (void)n_in; (void)out_size; (void)ws_size;

  const float* pair = (const float*)d_in[0];
  const unsigned char* mask = (const unsigned char*)d_in[1];
  const float* ln_w = (const float*)d_in[2];
  const float* ln_b = (const float*)d_in[3];
  const float* w_bias = (const float*)d_in[4];
  const float* w_q = (const float*)d_in[5];
  const float* w_k = (const float*)d_in[6];
  const float* w_v = (const float*)d_in[7];
  const float* w_g = (const float*)d_in[8];
  const float* w_o = (const float*)d_in[9];
  float* out = (float*)d_out;

  char* ws = (char*)d_ws;
  size_t off = 0;
  auto alloc = [&](size_t bytes) {
    size_t o = off;
    off += (bytes + 255) & ~(size_t)255;
    return o;
  };
  __bf16* wcat  = (__bf16*)(ws + alloc((size_t)528 * 128 * 2));
  __bf16* wobf  = (__bf16*)(ws + alloc((size_t)128 * 128 * 2));
  __bf16* xbf   = (__bf16*)(ws + alloc((size_t)NN * 128 * 2));
  __bf16* xp    = (__bf16*)(ws + alloc((size_t)NN * 384 * 2));
  __bf16* vt    = (__bf16*)(ws + alloc((size_t)NDIM * 4 * 32 * 384 * 2));
  float*  biasb = (float*) (ws + alloc((size_t)4 * NN * 4));
  __bf16* wabuf = (__bf16*)(ws + alloc((size_t)NN * 128 * 2));

  prep_weights<<<328, 256, 0, stream>>>(w_q, w_k, w_v, w_g, w_bias, w_o, wcat, wobf);
  ln_kernel<<<NN / 8, 256, 0, stream>>>(pair, ln_w, ln_b, xbf);
  proj_gemm<<<1152, 256, 528 * 136 * 2, stream>>>(xbf, wcat, xp, vt, biasb);
  attn_kernel<<<dim3(24, 384), 128, 0, stream>>>(xp, vt, biasb, mask, wabuf);
  out_gemm<<<1152, 256, 0, stream>>>(wabuf, wobf, out);
}
